// RateMatrixOutputHead_8976481649142
// MI455X (gfx1250) — matrix operand order repaired and checked by reading, not compiled, not run on hardware
//
#include <hip/hip_runtime.h>
#include <hip/hip_bf16.h>
#include <math.h>

typedef __bf16 bf16;
typedef __attribute__((ext_vector_type(16))) __bf16 v16bf;
typedef __attribute__((ext_vector_type(8)))  __bf16 v8bf;
typedef __attribute__((ext_vector_type(8)))  float  v8f;

#define MROWS 32768          // B*L
#define EDIM  1280
#define VDIM  32
#define PDIM  496
#define NC    528            // VDIM + PDIM
#define KT    40             // EDIM/32 k-tiles
#define NT1   80             // EDIM/16 n-tiles (dense gemm)
#define NT2   33             // NC/16 n-tiles (head gemm), 33*16 = 528

// ---------------------------------------------------------------------------
// Pack fp32 weights (K x N, row-major) into bf16 WMMA B-fragment lane order:
// tile (kt, nt) -> 32 lanes x 16 bf16 contiguous. Lane l: N = nt*16+(l&15),
// elems i=0..15 hold K = kt*32 + (l>>4)*8 + (i&7) + (i>>3)*16  (VGPR j = K pair 2j,2j+1).
// ---------------------------------------------------------------------------
__global__ void pack_w1(const float* __restrict__ W, bf16* __restrict__ out) {
  int t = blockIdx.x * blockDim.x + threadIdx.x;
  if (t >= KT * NT1 * 32) return;
  int lane = t & 31;
  int tile = t >> 5;
  int nt = tile % NT1;
  int kt = tile / NT1;
  int n  = nt * 16 + (lane & 15);
  int kh = (lane >> 4) * 8;
  bf16* dst = out + ((size_t)tile * 32 + lane) * 16;
#pragma unroll
  for (int i = 0; i < 16; ++i) {
    int k = kt * 32 + kh + (i & 7) + (i >> 3) * 16;
    dst[i] = (bf16)W[(size_t)k * EDIM + n];
  }
}

__global__ void pack_w2(const float* __restrict__ Wth,   // E x 32
                        const float* __restrict__ WTh,   // E x 496
                        bf16* __restrict__ out) {
  int t = blockIdx.x * blockDim.x + threadIdx.x;
  if (t >= KT * NT2 * 32) return;
  int lane = t & 31;
  int tile = t >> 5;
  int nt = tile % NT2;
  int kt = tile / NT2;
  int n  = nt * 16 + (lane & 15);
  int kh = (lane >> 4) * 8;
  bf16* dst = out + ((size_t)tile * 32 + lane) * 16;
#pragma unroll
  for (int i = 0; i < 16; ++i) {
    int k = kt * 32 + kh + (i & 7) + (i >> 3) * 16;
    float v = (n < VDIM) ? Wth[(size_t)k * VDIM + n]
                         : WTh[(size_t)k * PDIM + (n - VDIM)];
    dst[i] = (bf16)v;
  }
}

// ---------------------------------------------------------------------------
// Kernel 1: h = LayerNorm(gelu(hx @ dense_w + dense_b)) -> bf16
// Block: 256 threads (8 waves). One 16-row M-tile per block; wave w owns
// N-tiles [w*10, w*10+10). K loop over 40 tiles of 32.
// Inner loop: burst-load all 10 B fragments into independent regs, then 10
// WMMAs -> staggered s_wait_loadcnt instead of load->wait0->wmma serialization.
// ---------------------------------------------------------------------------
__global__ __launch_bounds__(256)
void dense_gelu_ln(const float* __restrict__ X, const bf16* __restrict__ Bp,
                   const float* __restrict__ bias, const float* __restrict__ gamma,
                   const float* __restrict__ beta, bf16* __restrict__ H) {
  const int lane  = threadIdx.x & 31;
  const int wv    = threadIdx.x >> 5;
  const int mbase = blockIdx.x * 16;
  const int rowA  = lane & 15;
  const int ksub  = (lane >> 4) * 8;
  const int half  = lane >> 4;

  v8f acc[10];
#pragma unroll
  for (int t = 0; t < 10; ++t) {
    v8f z = {0.f,0.f,0.f,0.f,0.f,0.f,0.f,0.f};
    acc[t] = z;
  }

  const float* pa0   = X + (size_t)(mbase + rowA) * EDIM + ksub;
  const v16bf* bbase = (const v16bf*)Bp + (size_t)(wv * 10) * 32 + lane;

  for (int kt = 0; kt < KT; ++kt) {
    const v16bf* bptr = bbase + (size_t)kt * (NT1 * 32);
    // burst-issue all B-fragment loads first (independent registers)
    v16bf b[10];
#pragma unroll
    for (int t = 0; t < 10; ++t) b[t] = bptr[(size_t)t * 32];

    // A fragment (16x32 bf16) from fp32 activations; cvt overlaps the loads.
    const float* pa = pa0 + kt * 32;
    if (kt + 1 < KT) {
      __builtin_prefetch(pa + 32, 0, 1);
      __builtin_prefetch(bptr + NT1 * 32, 0, 1);
    }
    float4 f0 = *(const float4*)(pa);
    float4 f1 = *(const float4*)(pa + 4);
    float4 f2 = *(const float4*)(pa + 16);
    float4 f3 = *(const float4*)(pa + 20);
    float af[16] = {f0.x,f0.y,f0.z,f0.w, f1.x,f1.y,f1.z,f1.w,
                    f2.x,f2.y,f2.z,f2.w, f3.x,f3.y,f3.z,f3.w};
    v16bf a;
#pragma unroll
    for (int i = 0; i < 16; ++i) a[i] = (bf16)af[i];

#pragma unroll
    for (int t = 0; t < 10; ++t) {
      acc[t] = __builtin_amdgcn_wmma_f32_16x16x32_bf16(
          false, a, false, b[t], (short)0, acc[t], false, false);
    }
  }

  // Epilogue: bias + exact-erf gelu, accumulate row stats.
  __shared__ float s_sum[16][8];
  __shared__ float s_sq[16][8];
  __shared__ float s_mu[16];
  __shared__ float s_rs[16];

  float ps[8], pq[8];
#pragma unroll
  for (int v = 0; v < 8; ++v) { ps[v] = 0.f; pq[v] = 0.f; }
#pragma unroll
  for (int t = 0; t < 10; ++t) {
    int ncol = (wv * 10 + t) * 16 + (lane & 15);
    float bb = bias[ncol];
#pragma unroll
    for (int v = 0; v < 8; ++v) {
      float x = acc[t][v] + bb;
      float g = 0.5f * x * (1.0f + erff(x * 0.70710678118654752f));
      acc[t][v] = g;
      ps[v] += g;
      pq[v] += g * g;
    }
  }
  // reduce across the 16-lane half that shares each row
#pragma unroll
  for (int v = 0; v < 8; ++v) {
#pragma unroll
    for (int off = 1; off < 16; off <<= 1) {
      ps[v] += __shfl_xor(ps[v], off, 32);
      pq[v] += __shfl_xor(pq[v], off, 32);
    }
  }
  if ((lane & 15) == 0) {
#pragma unroll
    for (int v = 0; v < 8; ++v) {
      s_sum[v + 8 * half][wv] = ps[v];
      s_sq [v + 8 * half][wv] = pq[v];
    }
  }
  __syncthreads();
  if (threadIdx.x < 16) {
    float s = 0.f, q = 0.f;
#pragma unroll
    for (int w = 0; w < 8; ++w) { s += s_sum[threadIdx.x][w]; q += s_sq[threadIdx.x][w]; }
    float mu  = s * (1.0f / EDIM);
    float var = q * (1.0f / EDIM) - mu * mu;
    s_mu[threadIdx.x] = mu;
    s_rs[threadIdx.x] = rsqrtf(var + 1e-12f);
  }
  __syncthreads();
#pragma unroll
  for (int t = 0; t < 10; ++t) {
    int ncol = (wv * 10 + t) * 16 + (lane & 15);
    float gm = gamma[ncol], bt = beta[ncol];
#pragma unroll
    for (int v = 0; v < 8; ++v) {
      int r = v + 8 * half;
      float h = (acc[t][v] - s_mu[r]) * s_rs[r] * gm + bt;
      H[(size_t)(mbase + r) * EDIM + ncol] = (bf16)h;
    }
  }
}

// ---------------------------------------------------------------------------
// Kernel 2: fused head. logits(16x528) = h @ [theta_w|Theta_w] + bias -> LDS,
// then per-row: masked log-softmax over V=32, softplus of 496 tri entries,
// Pande transform, row-sum correction. Writes Q and pi.
// ---------------------------------------------------------------------------
__global__ __launch_bounds__(256)
void head_fused(const bf16* __restrict__ H, const bf16* __restrict__ Bp,
                const float* __restrict__ theta_b, const float* __restrict__ Theta_b,
                const unsigned char* __restrict__ maskp,
                float* __restrict__ Qout, float* __restrict__ PiOut) {
  __shared__ float s_lg[16][NC + 4];
  const int lane  = threadIdx.x & 31;
  const int wv    = threadIdx.x >> 5;
  const int mbase = blockIdx.x * 16;
  const int rowA  = lane & 15;
  const int ksub  = (lane >> 4) * 8;
  const int half  = lane >> 4;

  v8f acc[5];
#pragma unroll
  for (int j = 0; j < 5; ++j) {
    v8f z = {0.f,0.f,0.f,0.f,0.f,0.f,0.f,0.f};
    acc[j] = z;
  }

  const bf16* pa0 = H + (size_t)(mbase + rowA) * EDIM + ksub;

  for (int kt = 0; kt < KT; ++kt) {
    // burst-load the (up to) 5 B fragments for this wave first
    v16bf b[5];
#pragma unroll
    for (int j = 0; j < 5; ++j) {
      int nt = wv + 8 * j;            // uniform per wave -> EXEC stays all-ones
      if (nt < NT2)
        b[j] = *((const v16bf*)Bp + ((size_t)kt * NT2 + nt) * 32 + lane);
    }
    const bf16* pa = pa0 + kt * 32;
    v8bf lo = *(const v8bf*)(pa);
    v8bf hi = *(const v8bf*)(pa + 16);
    v16bf a = __builtin_shufflevector(lo, hi, 0,1,2,3,4,5,6,7,8,9,10,11,12,13,14,15);
#pragma unroll
    for (int j = 0; j < 5; ++j) {
      int nt = wv + 8 * j;
      if (nt < NT2) {
        acc[j] = __builtin_amdgcn_wmma_f32_16x16x32_bf16(
            false, a, false, b[j], (short)0, acc[j], false, false);
      }
    }
  }

#pragma unroll
  for (int j = 0; j < 5; ++j) {
    int nt = wv + 8 * j;
    if (nt < NT2) {
      int ncol = nt * 16 + (lane & 15);
      float bb = (ncol < VDIM) ? theta_b[ncol] : Theta_b[ncol - VDIM];
#pragma unroll
      for (int v = 0; v < 8; ++v) s_lg[v + 8 * half][ncol] = acc[j][v] + bb;
    }
  }
  __syncthreads();

  int mval = maskp[lane] ? 1 : 0;    // lane == token index j
  for (int rr = 0; rr < 2; ++rr) {
    int r = wv + 8 * rr;
    int m = mbase + r;
    float lg = s_lg[r][lane];
    float ml = mval ? lg : -1e30f;
    float mx = ml;
#pragma unroll
    for (int off = 1; off < 32; off <<= 1) mx = fmaxf(mx, __shfl_xor(mx, off, 32));
    float ex = mval ? expf(ml - mx) : 0.0f;
    float se = ex;
#pragma unroll
    for (int off = 1; off < 32; off <<= 1) se += __shfl_xor(se, off, 32);
    float logpi = ml - mx - logf(se);
    float pi = mval ? expf(logpi) : 0.0f;
    PiOut[(size_t)m * VDIM + lane] = pi;
    float lps = mval ? logpi : 0.0f;   // log_pi_safe
    float e = expf(0.5f * lps);        // sqrt(pi_j)

    for (int i = 0; i < VDIM; ++i) {
      float ei = __shfl(e, i, 32);
      int   mi = __shfl(mval, i, 32);
      float s = 0.0f;
      if (i != lane) {
        int a = i < lane ? i : lane;
        int b = i < lane ? lane : i;
        int idx = a * (VDIM - 1) - (a * (a - 1)) / 2 + (b - a - 1);
        float t = s_lg[r][VDIM + idx];
        s = (t > 20.0f) ? t : log1pf(expf(t));   // softplus
      }
      if (!(mval && mi)) s = 0.0f;               // mask2d
      float q = s * (e / ei);                    // S_ij * sqrt(pi_j/pi_i)
      float rs = q;
#pragma unroll
      for (int off = 1; off < 32; off <<= 1) rs += __shfl_xor(rs, off, 32);
      float outv = (i == lane) ? -rs : q;        // diag row-sum correction
      Qout[(size_t)m * (VDIM * VDIM) + i * VDIM + lane] = outv;
    }
  }
}

// ---------------------------------------------------------------------------
extern "C" void kernel_launch(void* const* d_in, const int* in_sizes, int n_in,
                              void* d_out, int out_size, void* d_ws, size_t ws_size,
                              hipStream_t stream) {
  (void)in_sizes; (void)n_in; (void)out_size; (void)ws_size;
  const float*         hx      = (const float*)d_in[0];
  const unsigned char* maskp   = (const unsigned char*)d_in[1];
  const float*         dense_w = (const float*)d_in[2];
  const float*         dense_b = (const float*)d_in[3];
  const float*         ln_g    = (const float*)d_in[4];
  const float*         ln_b    = (const float*)d_in[5];
  const float*         theta_w = (const float*)d_in[6];
  const float*         theta_b = (const float*)d_in[7];
  const float*         Theta_w = (const float*)d_in[8];
  const float*         Theta_b = (const float*)d_in[9];

  char* ws = (char*)d_ws;
  bf16* hbuf = (bf16*)ws;
  size_t off = (size_t)MROWS * EDIM * sizeof(bf16);          // 83,886,080 B
  bf16* bpk1 = (bf16*)(ws + off);
  off += (size_t)KT * NT1 * 512 * sizeof(bf16);              // 3,276,800 B
  bf16* bpk2 = (bf16*)(ws + off);                            // 1,351,680 B

  float* Qout  = (float*)d_out;
  float* PiOut = Qout + (size_t)MROWS * VDIM * VDIM;

  {
    int tot = KT * NT1 * 32;
    pack_w1<<<(tot + 255) / 256, 256, 0, stream>>>(dense_w, bpk1);
  }
  {
    int tot = KT * NT2 * 32;
    pack_w2<<<(tot + 255) / 256, 256, 0, stream>>>(theta_w, Theta_w, bpk2);
  }
  dense_gelu_ln<<<MROWS / 16, 256, 0, stream>>>(hx, bpk1, dense_b, ln_g, ln_b, hbuf);
  head_fused<<<MROWS / 16, 256, 0, stream>>>(hbuf, bpk2, theta_b, Theta_b,
                                             maskp, Qout, PiOut);
}
